// BigBirdAttention_43825846288476
// MI455X (gfx1250) — matrix operand order repaired and checked by reading, not compiled, not run on hardware
//
#include <hip/hip_runtime.h>

// ============================================================================
// BigBird attention for MI455X (gfx1250, wave32, WMMA).
// Precision: bf16 inputs / fp32 accumulate via v_wmma_f32_16x16x32_bf16.
// Round 2 changes:
//   * attention K/V tiles staged in LDS once per 4-wave block (was 4x redundant
//     global loads) using gfx1250 async-to-LDS copies (ASYNCcnt) w/ fallback
//   * GEMM loops preload next A tile to overlap VMEM with WMMA
//   * softmax in log2 domain (exp2f -> single v_exp_f32)
// Workspace layout (needs 48MB of d_ws):
//   [0,8M)   xb   : x bf16              [4096,1024]
//   [8,10M)  Wqb, [10,12M) Wkb, [12,14M) Wvb, [14,16M) Wob   [1024,1024]
//   [16,24M) Qb   : Q bf16              [4096,1024]
//   [24,32M) Kb   : K bf16              [4096,1024]
//   [32,40M) Vt   : V transposed bf16   [16][64][4096]  (head, d, seq)
//   [40,48M) AO   : attn out bf16       [4096,1024]
// ============================================================================

#define S_LEN   4096
#define DMODEL  1024
#define NHEADS  16
#define DHEAD   64

#ifndef USE_ASYNC_LDS
#define USE_ASYNC_LDS 1
#endif

typedef __bf16 bf16;
typedef __attribute__((ext_vector_type(16))) __bf16 v16bf;
typedef __attribute__((ext_vector_type(8)))  __bf16 v8bf;
typedef __attribute__((ext_vector_type(4)))  __bf16 v4bf;
typedef __attribute__((ext_vector_type(8)))  float  v8f;
typedef __attribute__((ext_vector_type(4)))  float  v4f;

// ---- WMMA operand loaders (ISA 7.12.2 per-lane layouts, wave32) -------------
// A (16x32 bf16, MxK): lane L<16 holds row M=L, K = {0..7, 16..23};
//                      lane L>=16 holds row M=L-16, K = {8..15, 24..31}.
__device__ __forceinline__ v16bf load_A(const bf16* p, int ld) {
  const int lane = threadIdx.x & 31;
  const int row  = lane & 15;
  const int half = lane >> 4;
  const bf16* b0 = p + (size_t)row * ld + half * 8;   // K = half*8 + 0..7
  union { v16bf v; v8bf h[2]; } u;
  u.h[0] = *(const v8bf*)b0;        // 16B load
  u.h[1] = *(const v8bf*)(b0 + 16); // K = 16 + half*8 + 0..7
  return u.v;
}

// B (32x16 bf16, KxN): lane L<16 holds col N=L, K = {0..7, 16..23};
//                      lane L>=16 holds col N=L-16, K = {8..15, 24..31}.
// `p` points at a row-major [N][ld] array whose row n is column n of B.
__device__ __forceinline__ v16bf load_B(const bf16* p, int ld) {
  const int lane = threadIdx.x & 31;
  const int col  = lane & 15;
  const int half = lane >> 4;
  const bf16* b0 = p + (size_t)col * ld + half * 8;   // K = half*8 + 0..7
  union { v16bf v; v8bf h[2]; } u;
  u.h[0] = *(const v8bf*)b0;
  u.h[1] = *(const v8bf*)(b0 + 16);
  return u.v;
}

__device__ __forceinline__ v8f wmma_bf16(v16bf a, v16bf b, v8f c) {
  return __builtin_amdgcn_wmma_f32_16x16x32_bf16(false, a, false, b,
                                                 (short)0, c, false, false);
}

#if USE_ASYNC_LDS
// gfx1250 async LDS copy: one wave-wide instruction moves 16B per lane,
// LDS dest address in a VGPR (wave-relative byte offset = low 32 bits of the
// generic pointer per the LDS aperture rule), global src in a VGPR pair.
// Tracked by ASYNCcnt (ISA 10.x / 15.18.3).
__device__ __forceinline__ void async_copy16(bf16* lds_dst, const bf16* gsrc) {
  unsigned loff = (unsigned)(unsigned long long)lds_dst;
  asm volatile("global_load_async_to_lds_b128 %0, %1, off"
               :: "v"(loff), "v"(gsrc) : "memory");
}
__device__ __forceinline__ void wait_async0() {
  asm volatile("s_wait_asynccnt 0x0" ::: "memory");
}
#endif

// ---- fp32 -> bf16 conversion ------------------------------------------------
__global__ void cvt_kernel(const float* __restrict__ src, bf16* __restrict__ dst,
                           int n4) {
  int i = blockIdx.x * blockDim.x + threadIdx.x;
  if (i >= n4) return;
  v4f f = ((const v4f*)src)[i];
  v4bf b;
  b.x = (bf16)f.x; b.y = (bf16)f.y; b.z = (bf16)f.z; b.w = (bf16)f.w;
  ((v4bf*)dst)[i] = b;
}

// ---- QKV projection: one wave computes a 16x64 tile of Q, K or V ------------
__global__ void __launch_bounds__(128)
qkv_kernel(const bf16* __restrict__ xb,
           const bf16* __restrict__ Wqb, const bf16* __restrict__ Wkb,
           const bf16* __restrict__ Wvb,
           const float* __restrict__ bq, const float* __restrict__ bk,
           const float* __restrict__ bv,
           bf16* __restrict__ Qb, bf16* __restrict__ Kb, bf16* __restrict__ Vt) {
  const int w  = threadIdx.x >> 5;
  const int r0 = (blockIdx.x * 4 + w) * 16;   // seq rows
  const int c0 = blockIdx.y * 64;             // model cols (head-aligned)
  const int z  = blockIdx.z;
  const bf16*  W    = (z == 0) ? Wqb : (z == 1) ? Wkb : Wvb;
  const float* bias = (z == 0) ? bq  : (z == 1) ? bk  : bv;

  v8f acc[4] = {};
  const bf16* arow = xb + (size_t)r0 * DMODEL;
  v16bf a = load_A(arow, DMODEL);             // k = 0 tile, preloaded
  for (int k = 0; k < DMODEL; k += 32) {
    v16bf b0 = load_B(W + (size_t)(c0 +  0) * DMODEL + k, DMODEL);
    v16bf b1 = load_B(W + (size_t)(c0 + 16) * DMODEL + k, DMODEL);
    v16bf b2 = load_B(W + (size_t)(c0 + 32) * DMODEL + k, DMODEL);
    v16bf b3 = load_B(W + (size_t)(c0 + 48) * DMODEL + k, DMODEL);
    v16bf an = (k + 32 < DMODEL) ? load_A(arow + k + 32, DMODEL) : a;
    acc[0] = wmma_bf16(a, b0, acc[0]);
    acc[1] = wmma_bf16(a, b1, acc[1]);
    acc[2] = wmma_bf16(a, b2, acc[2]);
    acc[3] = wmma_bf16(a, b3, acc[3]);
    a = an;
  }

  const int lane = threadIdx.x & 31;
  const int col  = lane & 15;
  const int rb   = (lane >> 4) * 8;           // C layout: VGPR r -> row rb+r
#pragma unroll
  for (int n = 0; n < 4; ++n) {
    const int   gc = c0 + n * 16 + col;
    const float bb = bias[gc];
    v8f c = acc[n];
#pragma unroll
    for (int r = 0; r < 8; ++r) c[r] += bb;
    if (z < 2) {
      bf16* out = (z == 0 ? Qb : Kb) + (size_t)r0 * DMODEL + gc;
#pragma unroll
      for (int r = 0; r < 8; ++r) out[(size_t)(rb + r) * DMODEL] = (bf16)c[r];
    } else {
      v8bf pk;                      // transposed store: one b128 per lane
#pragma unroll
      for (int r = 0; r < 8; ++r) pk[r] = (bf16)c[r];
      *(v8bf*)(Vt + (size_t)gc * S_LEN + r0 + rb) = pk;
    }
  }
}

// ---- Flash attention: 4 waves/block share head h; K/V staged in LDS ---------
__global__ void __launch_bounds__(128)
attn_kernel(const bf16* __restrict__ Qb, const bf16* __restrict__ Kb,
            const bf16* __restrict__ Vt, const float* __restrict__ mask,
            bf16* __restrict__ AO) {
  __shared__ __align__(16) bf16 Kl[32 * 64];        // [j 0..31][d 0..63]
  __shared__ __align__(16) bf16 Vl[64 * 32];        // [d 0..63][j 0..31]
  __shared__ __align__(16) bf16 plds[4][16 * 32];   // per-wave P transpose
  const int t    = threadIdx.x;
  const int w    = t >> 5;
  const int q0   = (blockIdx.x * 4 + w) * 16;
  const int h    = blockIdx.y;
  const int lane = t & 31;
  const int col  = lane & 15;
  const int rb   = (lane >> 4) * 8;

  const bf16* Qh = Qb + (size_t)q0 * DMODEL + h * DHEAD;
  const v16bf qa0 = load_A(Qh, DMODEL);        // d = 0..31
  const v16bf qa1 = load_A(Qh + 32, DMODEL);   // d = 32..63

  float mrun[8], lrun[8];
#pragma unroll
  for (int r = 0; r < 8; ++r) { mrun[r] = -1e30f; lrun[r] = 0.0f; }
  v8f O[4] = {};

  const float l2e    = 1.44269504f;            // log2(e): softmax in 2^x domain
  const float scale2 = 0.125f * l2e;           // (1/sqrt(64)) * log2(e)
  const bf16* Kh  = Kb + h * DHEAD;
  const bf16* VtH = Vt + (size_t)h * DHEAD * S_LEN;
  bf16* lp = plds[w];

  for (int jb = 0; jb < S_LEN; jb += 32) {
    __syncthreads();   // all waves done reading previous Kl/Vl
    // ---- stage K tile (32x64) and Vt tile (64x32) into LDS ----
    // 256 + 256 chunks of 16B split across 128 threads.
#pragma unroll
    for (int c = 0; c < 2; ++c) {              // K: 256 chunks (8 per row)
      const int ch = t + c * 128;
      const int row = ch >> 3, off = (ch & 7) * 8;
      bf16*       d = Kl + row * 64 + off;
      const bf16* s = Kh + (size_t)(jb + row) * DMODEL + off;
#if USE_ASYNC_LDS
      async_copy16(d, s);
#else
      *(v8bf*)d = *(const v8bf*)s;
#endif
    }
#pragma unroll
    for (int c = 0; c < 2; ++c) {              // V: 256 chunks (4 per row)
      const int ch = t + c * 128;
      const int row = ch >> 2, off = (ch & 3) * 8;
      bf16*       d = Vl + row * 32 + off;
      const bf16* s = VtH + (size_t)row * S_LEN + jb + off;
#if USE_ASYNC_LDS
      async_copy16(d, s);
#else
      *(v8bf*)d = *(const v8bf*)s;
#endif
    }
#if USE_ASYNC_LDS
    wait_async0();
#endif
    __syncthreads();   // tiles visible to all waves

    // ---- scores: S = Q K^T (two 16x16 tiles over 32 j-columns) ----
    v8f s0 = {}, s1 = {};
    s0 = wmma_bf16(qa0, load_B(Kl, 64), s0);
    s0 = wmma_bf16(qa1, load_B(Kl + 32, 64), s0);
    s1 = wmma_bf16(qa0, load_B(Kl + 16 * 64, 64), s1);
    s1 = wmma_bf16(qa1, load_B(Kl + 16 * 64 + 32, 64), s1);

    // ---- scale + additive mask, logits in log2 domain ----
    float p0[8], p1[8], tmx[8];
    const float* mrow = mask + (size_t)q0 * S_LEN + jb;
#pragma unroll
    for (int r = 0; r < 8; ++r) {
      const size_t off = (size_t)(rb + r) * S_LEN;
      p0[r]  = s0[r] * scale2 + mrow[off + col] * l2e;
      p1[r]  = s1[r] * scale2 + mrow[off + col + 16] * l2e;
      tmx[r] = fmaxf(p0[r], p1[r]);
    }
    // ---- online softmax: row reductions across 16-lane halves ----
#pragma unroll
    for (int xm = 1; xm < 16; xm <<= 1)
#pragma unroll
      for (int r = 0; r < 8; ++r)
        tmx[r] = fmaxf(tmx[r], __shfl_xor(tmx[r], xm, 32));
    float corr[8], rs[8];
#pragma unroll
    for (int r = 0; r < 8; ++r) {
      const float nm = fmaxf(mrun[r], tmx[r]);
      corr[r] = exp2f(mrun[r] - nm);
      mrun[r] = nm;
      p0[r] = exp2f(p0[r] - nm);
      p1[r] = exp2f(p1[r] - nm);
      rs[r] = p0[r] + p1[r];
    }
#pragma unroll
    for (int xm = 1; xm < 16; xm <<= 1)
#pragma unroll
      for (int r = 0; r < 8; ++r)
        rs[r] += __shfl_xor(rs[r], xm, 32);
#pragma unroll
    for (int r = 0; r < 8; ++r) lrun[r] = lrun[r] * corr[r] + rs[r];
#pragma unroll
    for (int n = 0; n < 4; ++n)
#pragma unroll
      for (int r = 0; r < 8; ++r) O[n][r] *= corr[r];

    // ---- transpose P (C layout -> A layout) via per-wave LDS tile ----
    // LDS ops are in-order within a wave; no barrier needed (private region).
#pragma unroll
    for (int r = 0; r < 8; ++r) {
      lp[(rb + r) * 32 + col]      = (bf16)p0[r];
      lp[(rb + r) * 32 + col + 16] = (bf16)p1[r];
    }
    v16bf pa = load_A(lp, 32);

    // ---- O += P V from the staged Vt tile ----
#pragma unroll
    for (int n = 0; n < 4; ++n) {
      v16bf vb = load_B(Vl + (n * 16) * 32, 32);
      O[n] = wmma_bf16(pa, vb, O[n]);
    }
  }

  // ---- normalize + store bf16 attention output ----
#pragma unroll
  for (int n = 0; n < 4; ++n) {
    bf16* out = AO + (size_t)q0 * DMODEL + h * DHEAD + n * 16 + col;
#pragma unroll
    for (int r = 0; r < 8; ++r)
      out[(size_t)(rb + r) * DMODEL] = (bf16)(O[n][r] / lrun[r]);
  }
}

// ---- Output projection: out = AO @ Wo^T + bo (fp32 output) ------------------
__global__ void __launch_bounds__(128)
proj_kernel(const bf16* __restrict__ AO, const bf16* __restrict__ Wob,
            const float* __restrict__ bo, float* __restrict__ out) {
  const int w  = threadIdx.x >> 5;
  const int r0 = (blockIdx.x * 4 + w) * 16;
  const int c0 = blockIdx.y * 64;

  v8f acc[4] = {};
  const bf16* arow = AO + (size_t)r0 * DMODEL;
  v16bf a = load_A(arow, DMODEL);
  for (int k = 0; k < DMODEL; k += 32) {
    v16bf b0 = load_B(Wob + (size_t)(c0 +  0) * DMODEL + k, DMODEL);
    v16bf b1 = load_B(Wob + (size_t)(c0 + 16) * DMODEL + k, DMODEL);
    v16bf b2 = load_B(Wob + (size_t)(c0 + 32) * DMODEL + k, DMODEL);
    v16bf b3 = load_B(Wob + (size_t)(c0 + 48) * DMODEL + k, DMODEL);
    v16bf an = (k + 32 < DMODEL) ? load_A(arow + k + 32, DMODEL) : a;
    acc[0] = wmma_bf16(a, b0, acc[0]);
    acc[1] = wmma_bf16(a, b1, acc[1]);
    acc[2] = wmma_bf16(a, b2, acc[2]);
    acc[3] = wmma_bf16(a, b3, acc[3]);
    a = an;
  }
  const int lane = threadIdx.x & 31;
  const int col  = lane & 15;
  const int rb   = (lane >> 4) * 8;
#pragma unroll
  for (int n = 0; n < 4; ++n) {
    const int   gc = c0 + n * 16 + col;
    const float bb = bo[gc];
    float* o = out + (size_t)r0 * DMODEL + gc;
#pragma unroll
    for (int r = 0; r < 8; ++r)
      o[(size_t)(rb + r) * DMODEL] = acc[n][r] + bb;
  }
}

// ============================================================================
extern "C" void kernel_launch(void* const* d_in, const int* in_sizes, int n_in,
                              void* d_out, int out_size, void* d_ws, size_t ws_size,
                              hipStream_t stream) {
  const float* x    = (const float*)d_in[0];
  const float* Wq   = (const float*)d_in[1];
  const float* bq   = (const float*)d_in[2];
  const float* Wk   = (const float*)d_in[3];
  const float* bk   = (const float*)d_in[4];
  const float* Wv   = (const float*)d_in[5];
  const float* bv   = (const float*)d_in[6];
  const float* Wo   = (const float*)d_in[7];
  const float* bo   = (const float*)d_in[8];
  const float* mask = (const float*)d_in[9];

  char* ws = (char*)d_ws;
  const size_t MB = 1024ull * 1024ull;
  bf16* xb  = (bf16*)(ws + 0 * MB);
  bf16* Wqb = (bf16*)(ws + 8 * MB);
  bf16* Wkb = (bf16*)(ws + 10 * MB);
  bf16* Wvb = (bf16*)(ws + 12 * MB);
  bf16* Wob = (bf16*)(ws + 14 * MB);
  bf16* Qb  = (bf16*)(ws + 16 * MB);
  bf16* Kb  = (bf16*)(ws + 24 * MB);
  bf16* Vt  = (bf16*)(ws + 32 * MB);
  bf16* AO  = (bf16*)(ws + 40 * MB);

  const int n4 = S_LEN * DMODEL / 4;
  const int w4 = DMODEL * DMODEL / 4;
  cvt_kernel<<<(n4 + 255) / 256, 256, 0, stream>>>(x, xb, n4);
  cvt_kernel<<<(w4 + 255) / 256, 256, 0, stream>>>(Wq, Wqb, w4);
  cvt_kernel<<<(w4 + 255) / 256, 256, 0, stream>>>(Wk, Wkb, w4);
  cvt_kernel<<<(w4 + 255) / 256, 256, 0, stream>>>(Wv, Wvb, w4);
  cvt_kernel<<<(w4 + 255) / 256, 256, 0, stream>>>(Wo, Wob, w4);

  qkv_kernel<<<dim3(S_LEN / 64, DMODEL / 64, 3), 128, 0, stream>>>(
      xb, Wqb, Wkb, Wvb, bq, bk, bv, Qb, Kb, Vt);

  attn_kernel<<<dim3(S_LEN / 64, NHEADS), 128, 0, stream>>>(
      Qb, Kb, Vt, mask, AO);

  proj_kernel<<<dim3(S_LEN / 64, DMODEL / 64), 128, 0, stream>>>(
      AO, Wob, bo, (float*)d_out);

  (void)in_sizes; (void)n_in; (void)out_size; (void)ws_size;
}
